// SwinLayer_49349174231427
// MI455X (gfx1250) — matrix operand order repaired and checked by reading, not compiled, not run on hardware
//
#include <hip/hip_runtime.h>
#include <hip/hip_bf16.h>
#include <math.h>

// ---------------------------------------------------------------------------
// Swin layer (B=8, L=4096, C=512, H=8, DH=64, W=128, MLP=2048, INNER=64)
// for MI455X / gfx1250: all matmuls through v_wmma_f32_16x16x32_f16 (wave32),
// LDS staging through GLOBAL_LOAD_ASYNC_TO_LDS_B128 (ASYNCcnt) when available.
// ---------------------------------------------------------------------------

typedef __attribute__((ext_vector_type(16))) _Float16 v16h;
typedef __attribute__((ext_vector_type(8)))  float    v8f;
typedef _Float16 half_t;

#define ROWS   32768   // B*L
#define CDIM   512
#define HHEADS 8
#define DHEAD  64
#define WWIN   128
#define MLPD   2048
#define INNERD 64
#define NWIN   256     // ROWS / WWIN
#define NBATCH 2048    // NWIN * HHEADS
#define SEQ    4096

#define BM 128
#define BN 128
#define BK 32
#define LSTR 40        // LDS row stride in halves (80B, 16B aligned, padded)

// ---- CDNA5 async global->LDS copy (ASYNCcnt), with safe fallback ----------
// Builtin signature (from hipcc diagnostic): the b128 variant takes
// (int4 addrspace(1)*, int4 addrspace(3)*, imm int offset, imm int cpol).
#if defined(__has_builtin)
#if __has_builtin(__builtin_amdgcn_global_load_async_to_lds_b128)
#define HAVE_ASYNC 1
typedef int v4i_t __attribute__((vector_size(16)));
typedef __attribute__((address_space(1))) v4i_t* gv4i_p;
typedef __attribute__((address_space(3))) v4i_t* sv4i_p;
#define ASYNC_COPY16(gp, lp)                                                   \
  __builtin_amdgcn_global_load_async_to_lds_b128((gv4i_p)(const void*)(gp),    \
                                                 (sv4i_p)(void*)(lp), 0, 0)
#if __has_builtin(__builtin_amdgcn_s_wait_asynccnt)
#define WAIT_ASYNC() __builtin_amdgcn_s_wait_asynccnt(0)
#else
#define WAIT_ASYNC() asm volatile("s_wait_asynccnt 0x0" ::: "memory")
#endif
#endif
#endif
#ifndef HAVE_ASYNC
#define HAVE_ASYNC 0
#endif

// ------------------------------- elementwise -------------------------------

__global__ void k_f32_to_f16(const float* __restrict__ src, half_t* __restrict__ dst, int n) {
  for (int i = blockIdx.x * blockDim.x + threadIdx.x; i < n; i += gridDim.x * blockDim.x)
    dst[i] = (half_t)src[i];
}

// One row (C=512) per block of 256 threads. If gamma!=null: mona mix
// out = LN(x)*gamma + x*gammax (LN already includes g,b); else out = LN(x).
__global__ __launch_bounds__(256) void k_layernorm(
    const float* __restrict__ x, const float* __restrict__ g, const float* __restrict__ b,
    const float* __restrict__ gamma, const float* __restrict__ gammax,
    half_t* __restrict__ outH) {
  const int row = blockIdx.x;
  const int tid = threadIdx.x;
  const float* xr = x + (size_t)row * CDIM;
  float v0 = xr[tid], v1 = xr[tid + 256];
  __shared__ float rs[256], rq[256];
  rs[tid] = v0 + v1;
  rq[tid] = v0 * v0 + v1 * v1;
  __syncthreads();
  for (int st = 128; st > 0; st >>= 1) {
    if (tid < st) { rs[tid] += rs[tid + st]; rq[tid] += rq[tid + st]; }
    __syncthreads();
  }
  const float mean = rs[0] * (1.0f / CDIM);
  const float var  = rq[0] * (1.0f / CDIM) - mean * mean;
  const float rstd = rsqrtf(var + 1e-5f);
  #pragma unroll
  for (int j = 0; j < 2; j++) {
    const int c = tid + j * 256;
    const float xv = (j == 0) ? v0 : v1;
    float ln = (xv - mean) * rstd * g[c] + b[c];
    float o  = gamma ? (ln * gamma[c] + xv * gammax[c]) : ln;
    outH[(size_t)row * CDIM + c] = (half_t)o;
  }
}

// [row, h*64+d] f16 -> [win*8+h][s][d] f16, with optional RoPE (pos = s in window).
__global__ void k_rope_pack(const half_t* __restrict__ src, half_t* __restrict__ dst,
                            int doRope, int total /* ROWS*256 pair units */) {
  for (int t = blockIdx.x * blockDim.x + threadIdx.x; t < total; t += gridDim.x * blockDim.x) {
    const int row = t >> 8;           // /256
    const int p   = t & 255;
    const int h   = p >> 5;           // head
    const int i   = p & 31;           // pair index, d = 2i
    const int s   = row & (WWIN - 1);
    const int win = row >> 7;
    const size_t si = (size_t)row * CDIM + h * DHEAD + 2 * i;
    float x1 = (float)src[si], x2 = (float)src[si + 1];
    float r1 = x1, r2 = x2;
    if (doRope) {
      const float inv_freq = powf(10000.0f, -((float)(2 * i) / (float)DHEAD));
      float sn, cs;
      sincosf((float)s * inv_freq, &sn, &cs);
      r1 = x1 * cs - x2 * sn;
      r2 = x1 * sn + x2 * cs;
    }
    const size_t di = ((size_t)(win * HHEADS + h) * WWIN + s) * DHEAD + 2 * i;
    dst[di]     = (half_t)r1;
    dst[di + 1] = (half_t)r2;
  }
}

// in-place row softmax over 128 keys + post-softmax relative-position bias.
__global__ __launch_bounds__(128) void k_softmax_bias(half_t* __restrict__ sc,
                                                      const float* __restrict__ bias_table) {
  const int row = blockIdx.x;             // NBATCH*128 rows
  const int t   = threadIdx.x;            // key index 0..127
  const int q   = row & (WWIN - 1);
  const int h   = (row >> 7) & (HHEADS - 1);
  const float v = (float)sc[(size_t)row * WWIN + t];
  __shared__ float red[128];
  red[t] = v; __syncthreads();
  for (int st = 64; st > 0; st >>= 1) { if (t < st) red[t] = fmaxf(red[t], red[t + st]); __syncthreads(); }
  const float mx = red[0]; __syncthreads();
  const float e = __expf(v - mx);
  red[t] = e; __syncthreads();
  for (int st = 64; st > 0; st >>= 1) { if (t < st) red[t] += red[t + st]; __syncthreads(); }
  const float sum = red[0];
  const float p = e / sum + bias_table[(q - t + WWIN - 1) * HHEADS + h];
  sc[(size_t)row * WWIN + t] = (half_t)p;
}

// [win*8+h][s][d] f16 -> [row][h*64+d] f16
__global__ void k_ctx_unpack(const half_t* __restrict__ src, half_t* __restrict__ dst, int total) {
  for (int t = blockIdx.x * blockDim.x + threadIdx.x; t < total; t += gridDim.x * blockDim.x) {
    const int row = t >> 9;               // /512
    const int c   = t & 511;
    const int h   = c >> 6, d = c & 63;
    const int win = row >> 7, s = row & 127;
    dst[t] = src[((size_t)(win * HHEADS + h) * WWIN + s) * DHEAD + d];
  }
}

// mona depthwise convs k=3,5,7 over L (per 4096-row batch), channels-last [row,64].
__global__ void k_mona_conv(const float* __restrict__ h1,
                            const float* __restrict__ c1w, const float* __restrict__ c1b,
                            const float* __restrict__ c2w, const float* __restrict__ c2b,
                            const float* __restrict__ c3w, const float* __restrict__ c3b,
                            float* __restrict__ ccf, half_t* __restrict__ cch, int total) {
  for (int t = blockIdx.x * blockDim.x + threadIdx.x; t < total; t += gridDim.x * blockDim.x) {
    const int row = t >> 6, ch = t & 63;
    const int pos = row & (SEQ - 1);
    const int base = row - pos;
    float a3 = 0.f, a5 = 0.f, a7 = 0.f;
    #pragma unroll
    for (int j = 0; j < 3; j++) {
      const int p = pos + j - 1;
      if (p >= 0 && p < SEQ) a3 += c1w[ch * 3 + j] * h1[((size_t)(base + p) << 6) + ch];
    }
    #pragma unroll
    for (int j = 0; j < 5; j++) {
      const int p = pos + j - 2;
      if (p >= 0 && p < SEQ) a5 += c2w[ch * 5 + j] * h1[((size_t)(base + p) << 6) + ch];
    }
    #pragma unroll
    for (int j = 0; j < 7; j++) {
      const int p = pos + j - 3;
      if (p >= 0 && p < SEQ) a7 += c3w[ch * 7 + j] * h1[((size_t)(base + p) << 6) + ch];
    }
    const float cc = (a3 + c1b[ch] + a5 + c2b[ch] + a7 + c3b[ch]) * (1.0f / 3.0f) + h1[t];
    ccf[t] = cc;
    cch[t] = (half_t)cc;
  }
}

// ------------------------------- WMMA GEMM ---------------------------------

__device__ __forceinline__ float gelu_exact(float x) {
  return 0.5f * x * (1.0f + erff(x * 0.70710678118654752f));
}

union FragH { v16h h; uint4 u[2]; };

// C[M,N] = act(alpha * A[M,K] @ B + bias[N] + resid[M,N]); A,B f16, accum f32.
// BT=false: B is [K,N] row-major. BT=true: B element (k,n) at Bm[n*ldb + k].
// Block tile 128x128x32, 256 threads = 8 waves (4 in M, 2 in N), each wave
// 32x64 = 8 x v_wmma_f32_16x16x32_f16 accumulators.
// REQUIREMENT (guaranteed by all call sites): M % 128 == 0, K % 32 == 0,
// N % 16 == 0; A, B physically readable for the full 128-wide N tile.
template <bool BT, int ACT>
__global__ __launch_bounds__(256) void k_wmma_gemm(
    const half_t* __restrict__ A, const half_t* __restrict__ Bm,
    const float* __restrict__ bias, const float* __restrict__ resid,
    float* __restrict__ outF, half_t* __restrict__ outH,
    int M, int N, int K, int lda, int ldb, int ldc,
    long long strideA, long long strideB, long long strideC, float alpha) {
  __shared__ alignas(16) half_t ldsA[BM * LSTR];
  __shared__ alignas(16) half_t ldsB[BN * LSTR];

  const int tid  = threadIdx.x;
  const int lane = tid & 31;
  const int wave = tid >> 5;
  const int wm   = wave & 3;   // wave row  (0..3) -> 32 rows
  const int wn   = wave >> 2;  // wave col  (0..1) -> 64 cols
  const int bRow = blockIdx.y * BM;
  const int bCol = blockIdx.x * BN;
  const int batch = blockIdx.z;

  const half_t* Ap = A  + (long long)batch * strideA;
  const half_t* Bp = Bm + (long long)batch * strideB;

  const v8f vzero = {0.f, 0.f, 0.f, 0.f, 0.f, 0.f, 0.f, 0.f};
  v8f acc[2][4];
  #pragma unroll
  for (int i = 0; i < 2; i++)
    #pragma unroll
    for (int j = 0; j < 4; j++) acc[i][j] = vzero;

  // cooperative-load coordinates
  const int aRow = tid >> 1;                 // 0..127
  const int aOff = (tid & 1) * 16;           // halves
  const int bK   = tid & 31;                 // non-transposed B
  const int bN0  = (tid >> 5) * 16;
  const int btN  = tid >> 1;                 // transposed B
  const int btOff = (tid & 1) * 16;

  for (int k0 = 0; k0 < K; k0 += BK) {
    // ---- A tile: [128 x 32] halves, row-major (M % 128 == 0 -> no guard) ----
    {
      const half_t* gp = Ap + (size_t)(bRow + aRow) * lda + k0 + aOff;
#if HAVE_ASYNC
      half_t* lp = ldsA + aRow * LSTR + aOff;
      ASYNC_COPY16(gp, lp);
      ASYNC_COPY16(gp + 8, lp + 8);
#else
      const uint4* p = (const uint4*)gp;
      uint4 d0 = p[0], d1 = p[1];
      uint4* q = (uint4*)(ldsA + aRow * LSTR + aOff);
      q[0] = d0; q[1] = d1;
#endif
      if (k0 + BK < K) __builtin_prefetch(gp + BK, 0, 1);
    }
    // ---- B tile: stored in LDS as [n][k] (N-major) ----
    if (BT) {
      // index clamp keeps lanes convergent; out-of-range columns never stored
      const int gn = bCol + btN;
      const int gnc = gn < N ? gn : N - 1;
      const half_t* gp = Bp + (size_t)gnc * ldb + k0 + btOff;
#if HAVE_ASYNC
      half_t* lp = ldsB + btN * LSTR + btOff;
      ASYNC_COPY16(gp, lp);
      ASYNC_COPY16(gp + 8, lp + 8);
#else
      const uint4* p = (const uint4*)gp;
      uint4 d0 = p[0], d1 = p[1];
      uint4* q = (uint4*)(ldsB + btN * LSTR + btOff);
      q[0] = d0; q[1] = d1;
#endif
      if (k0 + BK < K) __builtin_prefetch(gp + BK, 0, 1);
    } else {
      // row-major weights need an N->K transpose into LDS; DMA cannot
      // transpose, so bounce through VGPRs and scatter 16-bit stores.
      const int gk = k0 + bK;
      const int gn0 = bCol + bN0;            // N is always a multiple of 16
      uint4 d0 = make_uint4(0, 0, 0, 0), d1 = d0;
      if (gn0 < N) {
        const uint4* p = (const uint4*)(Bp + (size_t)gk * ldb + gn0);
        d0 = p[0]; d1 = p[1];
        if (k0 + BK < K) __builtin_prefetch(Bp + (size_t)(gk + BK) * ldb + gn0, 0, 1);
      }
      union { uint4 u[2]; half_t hh[16]; } uu;
      uu.u[0] = d0; uu.u[1] = d1;
      #pragma unroll
      for (int j = 0; j < 16; j++) ldsB[(bN0 + j) * LSTR + bK] = uu.hh[j];
    }
#if HAVE_ASYNC
    WAIT_ASYNC();
#endif
    __syncthreads();

    // ---- fragments per ISA 16-bit A/B layouts ----
    FragH aF[2], bF[4];
    #pragma unroll
    for (int mt = 0; mt < 2; mt++) {
      // A: lane row = lane%16; VGPR0..3 hold K = 8*(lane/16)..+7, VGPR4..7 hold +16
      const half_t* pa = ldsA + (wm * 32 + mt * 16 + (lane & 15)) * LSTR + 8 * (lane >> 4);
      aF[mt].u[0] = *(const uint4*)pa;
      aF[mt].u[1] = *(const uint4*)(pa + 16);
    }
    #pragma unroll
    for (int nt = 0; nt < 4; nt++) {
      // B: lane col = lane%16; VGPR0..3 hold K = 8*(lane/16)..+7, VGPR4..7 hold +16
      const half_t* pb = ldsB + (wn * 64 + nt * 16 + (lane & 15)) * LSTR + 8 * (lane >> 4);
      bF[nt].u[0] = *(const uint4*)pb;
      bF[nt].u[1] = *(const uint4*)(pb + 16);
    }
    #pragma unroll
    for (int mt = 0; mt < 2; mt++)
      #pragma unroll
      for (int nt = 0; nt < 4; nt++)
        acc[mt][nt] = __builtin_amdgcn_wmma_f32_16x16x32_f16(
            false, aF[mt].h, false, bF[nt].h, (short)0, acc[mt][nt], false, false);
    __syncthreads();
  }

  // ---- epilogue: C/D layout: lane -> n = lane%16, m = v + 8*(lane/16) ----
  float*  outFb  = outF  ? outF  + (long long)batch * strideC : nullptr;
  half_t* outHb  = outH  ? outH  + (long long)batch * strideC : nullptr;
  const float* residb = resid ? resid + (long long)batch * strideC : nullptr;

  #pragma unroll
  for (int mt = 0; mt < 2; mt++) {
    #pragma unroll
    for (int nt = 0; nt < 4; nt++) {
      const int gn = bCol + wn * 64 + nt * 16 + (lane & 15);
      if (gn < N) {
        #pragma unroll
        for (int v = 0; v < 8; v++) {
          const int gm = bRow + wm * 32 + mt * 16 + v + 8 * (lane >> 4);  // < M always
          float val = acc[mt][nt][v] * alpha;
          if (bias)   val += bias[gn];
          if (residb) val += residb[(size_t)gm * ldc + gn];
          if (ACT == 1) val = gelu_exact(val);
          if (outFb) outFb[(size_t)gm * ldc + gn] = val;
          if (outHb) outHb[(size_t)gm * ldc + gn] = (half_t)val;
        }
      }
    }
  }
}

// ------------------------------- host side ---------------------------------

static void gemm(hipStream_t s, const half_t* A, const half_t* B, const float* bias,
                 const float* resid, float* outF, half_t* outH,
                 int M, int N, int K, int lda, int ldb, int ldc,
                 long long sA, long long sB, long long sC, int batch,
                 float alpha, bool bt, int act) {
  dim3 g((N + BN - 1) / BN, (M + BM - 1) / BM, batch);
  dim3 b(256);
  if (bt) {
    if (act) k_wmma_gemm<true, 1><<<g, b, 0, s>>>(A, B, bias, resid, outF, outH, M, N, K, lda, ldb, ldc, sA, sB, sC, alpha);
    else     k_wmma_gemm<true, 0><<<g, b, 0, s>>>(A, B, bias, resid, outF, outH, M, N, K, lda, ldb, ldc, sA, sB, sC, alpha);
  } else {
    if (act) k_wmma_gemm<false, 1><<<g, b, 0, s>>>(A, B, bias, resid, outF, outH, M, N, K, lda, ldb, ldc, sA, sB, sC, alpha);
    else     k_wmma_gemm<false, 0><<<g, b, 0, s>>>(A, B, bias, resid, outF, outH, M, N, K, lda, ldb, ldc, sA, sB, sC, alpha);
  }
}

static void cvt(hipStream_t s, const float* src, half_t* dst, int n) {
  k_f32_to_f16<<<(n + 255) / 256, 256, 0, s>>>(src, dst, n);
}

struct MonaIn {  // d_in indices for one mona block
  const float *norm_g, *norm_b, *gamma, *gammax, *b1, *b2;
  const float *c1w, *c1b, *c2w, *c2b, *c3w, *c3b, *pjb;
  const float *w1, *w2, *pjw;
};

extern "C" void kernel_launch(void* const* d_in, const int* in_sizes, int n_in,
                              void* d_out, int out_size, void* d_ws, size_t ws_size,
                              hipStream_t stream) {
  auto F = [&](int i) { return (const float*)d_in[i]; };
  // d_in order: 0 hidden, 1 ln1_g, 2 ln1_b, 3 ln2_g, 4 ln2_b, 5 wq, 6 bq, 7 wk,
  // 8 bk, 9 wv, 10 bv, 11 bias_table, 12 wo, 13 bo, 14 wi, 15 bi, 16 wo2, 17 bo2,
  // 18..33 mona1{norm_g,norm_b,gamma,gammax,w1,b1,w2,b2,c1_w,c1_b,c2_w,c2_b,c3_w,c3_b,pj_w,pj_b},
  // 34..49 mona2{...}
  const float* x_in = F(0);
  auto monaIn = [&](int base) {
    MonaIn m;
    m.norm_g = F(base + 0); m.norm_b = F(base + 1); m.gamma = F(base + 2); m.gammax = F(base + 3);
    m.w1 = F(base + 4); m.b1 = F(base + 5); m.w2 = F(base + 6); m.b2 = F(base + 7);
    m.c1w = F(base + 8); m.c1b = F(base + 9); m.c2w = F(base + 10); m.c2b = F(base + 11);
    m.c3w = F(base + 12); m.c3b = F(base + 13); m.pjw = F(base + 14); m.pjb = F(base + 15);
    return m;
  };
  MonaIn m1 = monaIn(18), m2 = monaIn(34);

  // workspace bump allocator (256B aligned)
  char* wsb = (char*)d_ws;
  size_t off = 0;
  auto alloc = [&](size_t bytes) {
    char* p = wsb + off;
    off = (off + bytes + 255) & ~(size_t)255;
    return p;
  };
  half_t* wq_h  = (half_t*)alloc((size_t)CDIM * CDIM * 2);
  half_t* wk_h  = (half_t*)alloc((size_t)CDIM * CDIM * 2);
  half_t* wv_h  = (half_t*)alloc((size_t)CDIM * CDIM * 2);
  half_t* wo_h  = (half_t*)alloc((size_t)CDIM * CDIM * 2);
  half_t* wi_h  = (half_t*)alloc((size_t)CDIM * MLPD * 2);
  half_t* wo2_h = (half_t*)alloc((size_t)MLPD * CDIM * 2);
  half_t* m1w1_h = (half_t*)alloc((size_t)CDIM * INNERD * 2);
  half_t* m1w2_h = (half_t*)alloc((size_t)INNERD * CDIM * 2);
  half_t* m1pj_h = (half_t*)alloc((size_t)INNERD * INNERD * 2);
  half_t* m2w1_h = (half_t*)alloc((size_t)CDIM * INNERD * 2);
  half_t* m2w2_h = (half_t*)alloc((size_t)INNERD * CDIM * 2);
  half_t* m2pj_h = (half_t*)alloc((size_t)INNERD * INNERD * 2);

  half_t* xn_h  = (half_t*)alloc((size_t)ROWS * CDIM * 2);
  half_t* q0    = (half_t*)alloc((size_t)ROWS * CDIM * 2);
  half_t* k0b   = (half_t*)alloc((size_t)ROWS * CDIM * 2);
  half_t* v0    = (half_t*)alloc((size_t)ROWS * CDIM * 2);
  half_t* qh    = (half_t*)alloc((size_t)ROWS * CDIM * 2);
  half_t* kh    = (half_t*)alloc((size_t)ROWS * CDIM * 2);
  half_t* vh    = (half_t*)alloc((size_t)ROWS * CDIM * 2);
  half_t* sc_h  = (half_t*)alloc((size_t)NBATCH * WWIN * WWIN * 2);  // scores->probs in-place
  half_t* ctx0  = (half_t*)alloc((size_t)NBATCH * WWIN * DHEAD * 2);
  half_t* ctxh  = (half_t*)alloc((size_t)ROWS * CDIM * 2);
  float*  hA    = (float*)alloc((size_t)ROWS * CDIM * 4);
  float*  hB    = (float*)alloc((size_t)ROWS * CDIM * 4);
  half_t* mid_h = (half_t*)alloc((size_t)ROWS * MLPD * 2);
  float*  h1f   = (float*)alloc((size_t)ROWS * INNERD * 4);
  float*  ccf   = (float*)alloc((size_t)ROWS * INNERD * 4);
  half_t* cch   = (half_t*)alloc((size_t)ROWS * INNERD * 2);
  half_t* gh    = (half_t*)alloc((size_t)ROWS * INNERD * 2);
  (void)ws_size; (void)in_sizes; (void)n_in; (void)out_size;

  // ---- 0: weights fp32 -> f16 ----
  cvt(stream, F(5),  wq_h,  CDIM * CDIM);
  cvt(stream, F(7),  wk_h,  CDIM * CDIM);
  cvt(stream, F(9),  wv_h,  CDIM * CDIM);
  cvt(stream, F(12), wo_h,  CDIM * CDIM);
  cvt(stream, F(14), wi_h,  CDIM * MLPD);
  cvt(stream, F(16), wo2_h, MLPD * CDIM);
  cvt(stream, m1.w1, m1w1_h, CDIM * INNERD);
  cvt(stream, m1.w2, m1w2_h, INNERD * CDIM);
  cvt(stream, m1.pjw, m1pj_h, INNERD * INNERD);
  cvt(stream, m2.w1, m2w1_h, CDIM * INNERD);
  cvt(stream, m2.w2, m2w2_h, INNERD * CDIM);
  cvt(stream, m2.pjw, m2pj_h, INNERD * INNERD);

  // ---- 1: LN1 -> xn_h ----
  k_layernorm<<<ROWS, 256, 0, stream>>>(x_in, F(1), F(2), nullptr, nullptr, xn_h);

  // ---- 2: QKV projections (WMMA) ----
  gemm(stream, xn_h, wq_h, F(6),  nullptr, nullptr, q0,  ROWS, CDIM, CDIM, CDIM, CDIM, CDIM, 0, 0, 0, 1, 1.0f, false, 0);
  gemm(stream, xn_h, wk_h, F(8),  nullptr, nullptr, k0b, ROWS, CDIM, CDIM, CDIM, CDIM, CDIM, 0, 0, 0, 1, 1.0f, false, 0);
  gemm(stream, xn_h, wv_h, F(10), nullptr, nullptr, v0,  ROWS, CDIM, CDIM, CDIM, CDIM, CDIM, 0, 0, 0, 1, 1.0f, false, 0);

  // ---- 3: RoPE + head-major repack ----
  const int pairTotal = ROWS * (CDIM / 2);
  k_rope_pack<<<(pairTotal + 255) / 256, 256, 0, stream>>>(q0,  qh, 1, pairTotal);
  k_rope_pack<<<(pairTotal + 255) / 256, 256, 0, stream>>>(k0b, kh, 1, pairTotal);
  k_rope_pack<<<(pairTotal + 255) / 256, 256, 0, stream>>>(v0,  vh, 0, pairTotal);

  // ---- 4: scores = Q K^T / sqrt(DH)  (batched, B transposed) ----
  gemm(stream, qh, kh, nullptr, nullptr, nullptr, sc_h,
       WWIN, WWIN, DHEAD, DHEAD, DHEAD, WWIN,
       (long long)WWIN * DHEAD, (long long)WWIN * DHEAD, (long long)WWIN * WWIN,
       NBATCH, 0.125f, true, 0);

  // ---- 5: softmax + post-softmax bias (in place) ----
  k_softmax_bias<<<NBATCH * WWIN, 128, 0, stream>>>(sc_h, F(11));

  // ---- 6: ctx = probs @ V (batched) ----
  gemm(stream, sc_h, vh, nullptr, nullptr, nullptr, ctx0,
       WWIN, DHEAD, WWIN, WWIN, DHEAD, DHEAD,
       (long long)WWIN * WWIN, (long long)WWIN * DHEAD, (long long)WWIN * DHEAD,
       NBATCH, 1.0f, false, 0);
  k_ctx_unpack<<<(ROWS * CDIM + 255) / 256, 256, 0, stream>>>(ctx0, ctxh, ROWS * CDIM);

  // ---- 7: output projection + residual -> hA ----
  gemm(stream, ctxh, wo_h, F(13), x_in, hA, nullptr, ROWS, CDIM, CDIM, CDIM, CDIM, CDIM, 0, 0, 0, 1, 1.0f, false, 0);

  // ---- mona adapter (xin -> xout) ----
  auto run_mona = [&](const MonaIn& m, const half_t* w1h, const half_t* w2h, const half_t* pjh,
                      const float* xin, float* xout) {
    k_layernorm<<<ROWS, 256, 0, stream>>>(xin, m.norm_g, m.norm_b, m.gamma, m.gammax, xn_h);
    gemm(stream, xn_h, w1h, m.b1, nullptr, h1f, nullptr,
         ROWS, INNERD, CDIM, CDIM, INNERD, INNERD, 0, 0, 0, 1, 1.0f, false, 0);
    k_mona_conv<<<(ROWS * INNERD + 255) / 256, 256, 0, stream>>>(
        h1f, m.c1w, m.c1b, m.c2w, m.c2b, m.c3w, m.c3b, ccf, cch, ROWS * INNERD);
    // pointwise conv (GEMM, B transposed) + residual cc + exact GELU -> gh
    gemm(stream, cch, pjh, m.pjb, ccf, nullptr, gh,
         ROWS, INNERD, INNERD, INNERD, INNERD, INNERD, 0, 0, 0, 1, 1.0f, true, 1);
    // identity + g @ w2 + b2 -> xout (f32)
    gemm(stream, gh, w2h, m.b2, xin, xout, nullptr,
         ROWS, CDIM, INNERD, INNERD, CDIM, CDIM, 0, 0, 0, 1, 1.0f, false, 0);
  };

  // ---- 8: mona1 (hA -> hB) ----
  run_mona(m1, m1w1_h, m1w2_h, m1pj_h, hA, hB);

  // ---- 9: MLP: LN2 -> gelu(x@wi+bi)@wo2+bo2 + residual -> hA ----
  k_layernorm<<<ROWS, 256, 0, stream>>>(hB, F(3), F(4), nullptr, nullptr, xn_h);
  gemm(stream, xn_h, wi_h, F(15), nullptr, nullptr, mid_h,
       ROWS, MLPD, CDIM, CDIM, MLPD, MLPD, 0, 0, 0, 1, 1.0f, false, 1);
  gemm(stream, mid_h, wo2_h, F(17), hB, hA, nullptr,
       ROWS, CDIM, MLPD, MLPD, CDIM, CDIM, 0, 0, 0, 1, 1.0f, false, 0);

  // ---- 10: mona2 (hA -> d_out, f32) ----
  run_mona(m2, m2w1_h, m2w2_h, m2pj_h, hA, (float*)d_out);
}
